// CGraphGNN_57088705298488
// MI455X (gfx1250) — matrix operand order repaired and checked by reading, not compiled, not run on hardware
//
#include <hip/hip_runtime.h>
#include <hip/hip_bf16.h>

typedef __attribute__((ext_vector_type(16))) _Float16 v16h;
typedef __attribute__((ext_vector_type(8)))  _Float16 v8h;
typedef __attribute__((ext_vector_type(8)))  float    v8f;

#define BB   32
#define NN   256
#define FF   12
#define EH   32
#define EOUT 30
#define NOUT 12
#define LOG2E 1.44269504088896f

// plain fast sigmoid (cold paths)
__device__ __forceinline__ float fsig(float x) {
    return __builtin_amdgcn_rcpf(1.f + __expf(-x));
}
// sigmoid of x where y = x*log2(e) is given (scale pre-folded into weights/bias):
// rcp(1 + exp2(-y)); negate is a free src modifier on v_exp_f32
__device__ __forceinline__ float fsig2(float y) {
    return __builtin_amdgcn_rcpf(1.f + __builtin_amdgcn_exp2f(-y));
}

// lane L <-> lane L^16 swap, pure VALU (v_permlanex16_b32, identity lane select)
__device__ __forceinline__ int xswap16_i(int v) {
    return __builtin_amdgcn_permlanex16(v, v, 0x76543210, 0xfedcba98, false, false);
}
__device__ __forceinline__ float xswap16_f(float v) {
    return __int_as_float(xswap16_i(__float_as_int(v)));
}

// ---------------- Pass 1: per-column stats of adj_dist / adj_direct ----------------
__global__ __launch_bounds__(256) void stats_partial_k(const float* __restrict__ dist,
                                                       const float* __restrict__ dirc,
                                                       float* __restrict__ sums) {
    int j  = threadIdx.x;          // column
    int r0 = blockIdx.x * 32;      // 8192 rows / 256 blocks
    float sd = 0.f, sd2 = 0.f, sr = 0.f, sr2 = 0.f;
    for (int r = r0; r < r0 + 32; ++r) {
        float d = dist[r * NN + j];
        float t = dirc[r * NN + j];
        sd += d; sd2 += d * d; sr += t; sr2 += t * t;
    }
    atomicAdd(&sums[j], sd);
    atomicAdd(&sums[NN + j], sd2);
    atomicAdd(&sums[2 * NN + j], sr);
    atomicAdd(&sums[3 * NN + j], sr2);
}

__global__ __launch_bounds__(256) void stats_final_k(const float* __restrict__ sums,
                                                     float* __restrict__ norm) {
    int j = threadIdx.x;
    const float n = (float)(BB * NN);
    float md = sums[j] / n;
    float vd = (sums[NN + j] - n * md * md) / (n - 1.f);     // ddof=1
    float mr = sums[2 * NN + j] / n;
    float vr = (sums[3 * NN + j] - n * mr * mr) / (n - 1.f);
    norm[j]          = md;
    norm[NN + j]     = rsqrtf(vd);
    norm[2 * NN + j] = mr;
    norm[3 * NN + j] = rsqrtf(vr);
}

// ---------------- Pass 2: fused edge MLP, transposed GEMMs, register-only fragments ----
// Layer 1:  h^T(32x16) = (log2e*W1^T) @ f^T + (log2e*b1)  -> bias rides in WMMA C operand
// Layer 2:  e^T(30x16) = (log2e*W2^T) @ h^T + (log2e*b2)  -> B built via permlanex16 swap
__global__ __launch_bounds__(256) void edge_mlp_k(
    const float* __restrict__ x,
    const float* __restrict__ adist,
    const float* __restrict__ adirc,
    const float* __restrict__ W1, const float* __restrict__ b1,
    const float* __restrict__ W2, const float* __restrict__ b2,
    const float* __restrict__ norm,
    float* __restrict__ agg) {

    __shared__ float xLDS[NN * FF];      // x[b] : 12 KB
    __shared__ float normL[4 * NN];      // per-j norm params : 4 KB
    __shared__ float inAcc[NN][EOUT];    // incoming accumulator : 30 KB

    const int b    = blockIdx.x >> 4;
    const int i0   = (blockIdx.x & 15) << 4;
    const int tid  = threadIdx.x;
    const int lane = tid & 31;
    const int wave = tid >> 5;
    const int c    = lane & 15;     // edge slot within tile / A-row index
    const int half = lane >> 4;
    const int mrow = half * 8;      // C/D row offset for this lane half

    for (int u = tid; u < NN * FF; u += 256)   xLDS[u]  = x[b * NN * FF + u];
    for (int u = tid; u < 4 * NN; u += 256)    normL[u] = norm[u];
    for (int u = tid; u < NN * EOUT; u += 256) (&inAcc[0][0])[u] = 0.f;
    __syncthreads();

    // ---- constant A-matrix fragments (pre-scaled by log2e):
    // A layout: lane<16 holds K {0..15}; lane>=16 holds K {16..31}; row M = lane%16
    v16h a1a, a1b, a2a, a2b;
    {
        const int m  = c;
        const int ka = half ? 16 : 0;
#pragma unroll
        for (int t = 0; t < 16; ++t) {
            const int k = ka + t;
            a1a[t] = (_Float16)(W1[k * EH + m] * LOG2E);
            a1b[t] = (_Float16)(W1[k * EH + m + 16] * LOG2E);
            a2a[t] = (_Float16)(W2[k * EOUT + m] * LOG2E);
            a2b[t] = (m + 16 < EOUT) ? (_Float16)(W2[k * EOUT + m + 16] * LOG2E)
                                     : (_Float16)0.f;
        }
    }
    // bias vectors, pre-scaled, used directly as WMMA C operand (row = channel)
    v8f b1a, b1b, b2a, b2b;
#pragma unroll
    for (int v = 0; v < 8; ++v) {
        b1a[v] = b1[mrow + v] * LOG2E;
        b1b[v] = b1[16 + mrow + v] * LOG2E;
        b2a[v] = b2[mrow + v] * LOG2E;
        b2b[v] = (16 + mrow + v < EOUT) ? b2[16 + mrow + v] * LOG2E : 0.f;
    }

    const v8f zacc = {0.f, 0.f, 0.f, 0.f, 0.f, 0.f, 0.f, 0.f};
    typedef union { v8h h8; int i4[4]; } pack8;

    for (int rep = 0; rep < 2; ++rep) {
        const int i = i0 + wave + rep * 8;            // source node
        const float* xs = &xLDS[i * FF];
        const float* distRow = adist + (size_t)(b * NN + i) * NN;
        const float* dircRow = adirc + (size_t)(b * NN + i) * NN;
        v8f oa = zacc, ob = zacc;                     // outgoing partial sums (register)

        for (int g = 0; g < 4; ++g) {                 // 4 groups x 4 tiles
            const int myj0 = g * 64 + 4 * c;          // this lane's 4 edges (b128 load)
            // half0 lanes fetch dist, half1 lanes fetch direct; exchange via permlanex16
            const float4 mine = half ? *(const float4*)(dircRow + myj0)
                                     : *(const float4*)(distRow + myj0);
            float dv[4], rv[4];
            {
                const float m4[4] = {mine.x, mine.y, mine.z, mine.w};
#pragma unroll
                for (int q = 0; q < 4; ++q) {
                    const float o = xswap16_f(m4[q]);   // other half's value
                    dv[q] = half ? o : m4[q];           // dist (both halves)
                    rv[q] = m4[q];                      // direct (meaningful on half1)
                }
            }

#pragma unroll
            for (int q = 0; q < 4; ++q) {
                const int j = myj0 + q;               // this lane's edge target
                const float d = dv[q];
                const float a = (d != 0.f) ? 1.f : 0.f;
                const float* xt = &xLDS[j * FF];

                // ---- B fragment of f^T built directly in registers:
                // lane = edge column c; lane half supplies K 0..15 / 16..31 contiguous
                v16h bf;
                if (half == 0) {
#pragma unroll
                    for (int k = 0; k < 12; ++k) bf[k] = (_Float16)(a * xs[k]);
                    bf[12] = (_Float16)a;
                    bf[13] = (_Float16)(a * xt[0]);
                    bf[14] = (_Float16)(a * xt[1]);
                    bf[15] = (_Float16)(a * xt[2]);
                } else {
                    const float dn = (d     - normL[j])          * normL[NN + j];
                    const float rn = (rv[q] - normL[2 * NN + j]) * normL[3 * NN + j];
#pragma unroll
                    for (int k = 0; k < 9; ++k)  bf[k] = (_Float16)(a * xt[3 + k]);
                    bf[9]  = (_Float16)a;
                    bf[10] = (_Float16)dn;
                    bf[11] = (_Float16)a;
                    bf[12] = (_Float16)rn;
                    bf[13] = (_Float16)a; bf[14] = (_Float16)a; bf[15] = (_Float16)a;
                }

                // ---- layer 1: h^T = W1'^T @ f^T + b1' (bias in C operand) ----
                v8f ha = __builtin_amdgcn_wmma_f32_16x16x32_f16(false, a1a, false, bf,
                                                                (short)0, b1a, false, false);
                v8f hb = __builtin_amdgcn_wmma_f32_16x16x32_f16(false, a1b, false, bf,
                                                                (short)0, b1b, false, false);

                // sigmoid (exp2-form), pack: plo = hidden {mrow..mrow+7}, phi = {16+mrow..}
                pack8 plo, phi;
#pragma unroll
                for (int v = 0; v < 8; ++v) {
                    plo.h8[v] = (_Float16)fsig2(ha[v]);
                    phi.h8[v] = (_Float16)fsig2(hb[v]);
                }
                // half-swap (VALU permlanex16) converts D K-split -> B K-split
                pack8 qlo, qhi;
#pragma unroll
                for (int t = 0; t < 4; ++t) {
                    qlo.i4[t] = xswap16_i(plo.i4[t]);
                    qhi.i4[t] = xswap16_i(phi.i4[t]);
                }
                v16h bh;
#pragma unroll
                for (int t = 0; t < 8; ++t) {
                    bh[t]     = (half == 0) ? plo.h8[t] : qhi.h8[t];  // K 0..7 | 16..23
                    bh[t + 8] = (half == 0) ? qlo.h8[t] : phi.h8[t];  // K 8..15 | 24..31
                }

                // ---- layer 2: e^T = W2'^T @ h^T + b2' ----
                v8f e2a = __builtin_amdgcn_wmma_f32_16x16x32_f16(false, a2a, false, bh,
                                                                 (short)0, b2a, false, false);
                v8f e2b = __builtin_amdgcn_wmma_f32_16x16x32_f16(false, a2b, false, bh,
                                                                 (short)0, b2b, false, false);

                // ---- aggregate: D column = edge j, D row = output channel ----
#pragma unroll
                for (int v = 0; v < 8; ++v) {
                    float ev = fsig2(e2a[v]);
                    atomicAdd(&inAcc[j][mrow + v], ev);   // incoming (sum over i)
                    oa[v] += ev;                           // outgoing partial (sum over j)
                    const int ch = 16 + mrow + v;
                    if (ch < EOUT) {
                        float ev2 = fsig2(e2b[v]);
                        atomicAdd(&inAcc[j][ch], ev2);
                        ob[v] += ev2;
                    }
                }
            }
        }

        // ---- outgoing: butterfly-reduce over the 16 edge lanes of each half ----
#pragma unroll
        for (int mask = 1; mask <= 8; mask <<= 1) {
#pragma unroll
            for (int v = 0; v < 8; ++v) {
                oa[v] += __shfl_xor(oa[v], mask, 32);
                ob[v] += __shfl_xor(ob[v], mask, 32);
            }
        }
        if (c == 0) {   // lanes 0 and 16: one writer per channel group
            float* gi = &agg[(b * NN + i) * EOUT];
#pragma unroll
            for (int v = 0; v < 8; ++v) atomicAdd(&gi[mrow + v], -oa[v]);
#pragma unroll
            for (int v = 0; v < 8; ++v) {
                const int ch = 16 + mrow + v;
                if (ch < EOUT) atomicAdd(&gi[ch], -ob[v]);
            }
        }
    }

    __syncthreads();
    // flush incoming accumulator (this block contributed 16 of 256 source rows)
    {
        const int j = tid;
        for (int cc = 0; cc < EOUT; ++cc)
            atomicAdd(&agg[(b * NN + j) * EOUT + cc], inAcc[j][cc]);
    }
}

// ---------------- Pass 3: node MLP 30 -> 12 ----------------
__global__ __launch_bounds__(256) void node_mlp_k(const float* __restrict__ agg,
                                                  const float* __restrict__ W3,
                                                  const float* __restrict__ b3,
                                                  float* __restrict__ out) {
    int t = blockIdx.x * 256 + threadIdx.x;     // 0 .. B*N-1
    const float* g = &agg[t * EOUT];
    float acc[NOUT];
#pragma unroll
    for (int cc = 0; cc < NOUT; ++cc) acc[cc] = b3[cc];
#pragma unroll
    for (int k = 0; k < EOUT; ++k) {
        float gv = g[k];
#pragma unroll
        for (int cc = 0; cc < NOUT; ++cc) acc[cc] += gv * W3[k * NOUT + cc];
    }
#pragma unroll
    for (int cc = 0; cc < NOUT; ++cc) out[t * NOUT + cc] = fsig(acc[cc]);
}

extern "C" void kernel_launch(void* const* d_in, const int* in_sizes, int n_in,
                              void* d_out, int out_size, void* d_ws, size_t ws_size,
                              hipStream_t stream) {
    (void)in_sizes; (void)n_in; (void)out_size; (void)ws_size;
    const float* x     = (const float*)d_in[0];
    const float* adist = (const float*)d_in[1];
    const float* adirc = (const float*)d_in[2];
    // d_in[3], d_in[4] (wind_mean / wind_std) unused by the reference
    const float* W1 = (const float*)d_in[5];
    const float* b1 = (const float*)d_in[6];
    const float* W2 = (const float*)d_in[7];
    const float* b2 = (const float*)d_in[8];
    const float* W3 = (const float*)d_in[9];
    const float* b3 = (const float*)d_in[10];

    float* ws   = (float*)d_ws;
    float* sums = ws;                 // 4*256 floats
    float* norm = ws + 1024;          // 4*256 floats
    float* agg  = ws + 2048;          // B*N*EOUT floats

    // zero stats + agg (capture-safe)
    hipMemsetAsync(d_ws, 0, (size_t)(2048 + BB * NN * EOUT) * sizeof(float), stream);

    stats_partial_k<<<256, 256, 0, stream>>>(adist, adirc, sums);
    stats_final_k<<<1, 256, 0, stream>>>(sums, norm);
    edge_mlp_k<<<BB * 16, 256, 0, stream>>>(x, adist, adirc, W1, b1, W2, b2, norm, agg);
    node_mlp_k<<<BB * NN / 256, 256, 0, stream>>>(agg, W3, b3, (float*)d_out);
}
